// PMPGNN_82119774699930
// MI455X (gfx1250) — matrix operand order repaired and checked by reading, not compiled, not run on hardware
//
#include <hip/hip_runtime.h>
#include <hip/hip_bf16.h>
#include <math.h>

#define N_NODES 50000
#define N_EDGES 800000
#define NFEAT   512
#define NHID    64
#define NCLASS  16

typedef __bf16 bf16_t;
typedef __attribute__((ext_vector_type(16))) __bf16 v16bf;
typedef __attribute__((ext_vector_type(8)))  float  v8f;

#ifndef __has_builtin
#define __has_builtin(x) 0
#endif
#if __has_builtin(__builtin_amdgcn_global_load_async_to_lds_b128) && \
    __has_builtin(__builtin_amdgcn_s_wait_asynccnt)
#define HAVE_ASYNC_LDS 1
typedef int v4i_async __attribute__((ext_vector_type(4)));
typedef __attribute__((address_space(1))) v4i_async* g4_ptr;   // global int4*
typedef __attribute__((address_space(3))) v4i_async* l4_ptr;   // LDS int4*
#else
#define HAVE_ASYNC_LDS 0
#endif

__device__ __forceinline__ float elu_f(float x)   { return x > 0.f ? x : (expf(x) - 1.f); }
__device__ __forceinline__ float lrelu_f(float x) { return x > 0.f ? x : 0.2f * x; }

__device__ __forceinline__ void atomicMaxF(float* addr, float val) {
    int* ai = (int*)addr;
    int old = __float_as_int(*addr);
    while (__int_as_float(old) < val) {
        int assumed = old;
        old = atomicCAS(ai, assumed, __float_as_int(val));
        if (old == assumed) break;
    }
}

// ---------------------------------------------------------------------------
// Weight prep: bf16 transposed copies (B^T[n][k]) so that a WMMA B fragment
// (lane = col N, K = 2p + 16*half, the k order this code picks) is 16 CONTIGUOUS bf16 = one 32B load.
// ---------------------------------------------------------------------------
__global__ void prep_weights_kernel(const float* __restrict__ W1, const float* __restrict__ W2,
                                    bf16_t* __restrict__ wt1, bf16_t* __restrict__ wt2)
{
    int i = blockIdx.x * blockDim.x + threadIdx.x;
    if (i < NFEAT * NHID) {                 // fc1_W [512,64] -> wt1 [64,512]
        int k = i >> 6, n = i & 63;
        wt1[(size_t)n * NFEAT + k] = (bf16_t)W1[i];
    }
    if (i < NHID * NCLASS) {                // fc2_W [64,16] -> wt2 [16,64]
        int k = i >> 4, n = i & 15;
        wt2[(size_t)n * NHID + k] = (bf16_t)W2[i];
    }
}

// ---------------------------------------------------------------------------
// Unified WMMA GEMM:  FS[N, NT*16] = X[N, K] @ W[K, NT*16]
// Block: 256 threads (8 waves), tile M=128. A tile (128x32 f32) staged in LDS
// (async DMA when available, double-buffered); B fragments read directly from
// the bf16 transposed weights in global (L2-resident).
// ---------------------------------------------------------------------------
#define LDAF 36   // f32 per A-row in LDS: 144B (16B aligned), 36-bank offset

template<int K>
__device__ __forceinline__ void stage_tile(const float* __restrict__ X, float* dstbuf,
                                           int t, int m0, int k0)
{
#pragma unroll
    for (int i = 0; i < 4; ++i) {
        int lin = t + i * 256;          // 1024 16B-slots of the 128x32 tile
        int r   = lin >> 3;
        int c4  = lin & 7;
        int gm  = m0 + r;
        if (gm > N_NODES - 1) gm = N_NODES - 1;   // clamp: results discarded later
        const float* gp = X + (size_t)gm * K + k0 + c4 * 4;
        float*       lp = dstbuf + r * LDAF + c4 * 4;
#if HAVE_ASYNC_LDS
        __builtin_amdgcn_global_load_async_to_lds_b128((g4_ptr)(void*)(const void*)gp,
                                                       (l4_ptr)(void*)lp, 0, 0);
#else
        *(float4*)lp = *(const float4*)gp;
#endif
    }
}

template<int K, int NT>
__global__ __launch_bounds__(256)
void gemm_wmma_kernel(const float* __restrict__ X, const bf16_t* __restrict__ Wt,
                      float* __restrict__ FS)
{
    __shared__ float Asf[2][128 * LDAF];

    const int t     = threadIdx.x;
    const int wave  = t >> 5;
    const int lane  = t & 31;
    const int half  = lane >> 4;
    const int row16 = lane & 15;
    const int m0    = blockIdx.x * 128;
    constexpr int NOUT = NT * 16;

    v8f acc[NT];
#pragma unroll
    for (int i = 0; i < NT; ++i)
#pragma unroll
        for (int j = 0; j < 8; ++j) acc[i][j] = 0.f;

    stage_tile<K>(X, Asf[0], t, m0, 0);

    for (int k0 = 0, it = 0; k0 < K; k0 += 32, ++it) {
        const int buf = it & 1;
#if HAVE_ASYNC_LDS
        __builtin_amdgcn_s_wait_asynccnt(0);
#endif
        __syncthreads();
        if (k0 + 32 < K) stage_tile<K>(X, Asf[buf ^ 1], t, m0, k0 + 32);

        // A fragment: row M = wave*16 + (lane&15);
        // K run: [16h .. 16h+15]  (64B, 16B aligned)
        const float* arow = &Asf[buf][(wave * 16 + row16) * LDAF];
        float4 a0 = *(const float4*)(arow + 16 * half);
        float4 a1 = *(const float4*)(arow + 16 * half + 4);
        float4 a2 = *(const float4*)(arow + 16 * half + 8);
        float4 a3 = *(const float4*)(arow + 16 * half + 12);
        v16bf af;
        af[0]  = (bf16_t)a0.x; af[1]  = (bf16_t)a0.y; af[2]  = (bf16_t)a0.z; af[3]  = (bf16_t)a0.w;
        af[4]  = (bf16_t)a1.x; af[5]  = (bf16_t)a1.y; af[6]  = (bf16_t)a1.z; af[7]  = (bf16_t)a1.w;
        af[8]  = (bf16_t)a2.x; af[9]  = (bf16_t)a2.y; af[10] = (bf16_t)a2.z; af[11] = (bf16_t)a2.w;
        af[12] = (bf16_t)a3.x; af[13] = (bf16_t)a3.y; af[14] = (bf16_t)a3.z; af[15] = (bf16_t)a3.w;

#pragma unroll
        for (int nt = 0; nt < NT; ++nt) {
            // B fragment: 16 contiguous bf16 at Wt[(nt*16+row16)*K + k0 + 16*half]
            const bf16_t* bp = Wt + (size_t)(nt * 16 + row16) * K + k0 + 16 * half;
            v16bf bfr = *(const v16bf*)bp;
            acc[nt] = __builtin_amdgcn_wmma_f32_16x16x32_bf16(
                false, af, false, bfr, (short)0, acc[nt], false, false);
        }
    }

    // D layout: VGPR v -> M = v + 8*half (within tile), N = lane&15
#pragma unroll
    for (int nt = 0; nt < NT; ++nt) {
#pragma unroll
        for (int v = 0; v < 8; ++v) {
            int m = m0 + wave * 16 + v + 8 * half;
            int n = nt * 16 + row16;
            if (m < N_NODES) FS[(size_t)m * NOUT + n] = acc[nt][v];
        }
    }
}

// ---------------------------------------------------------------------------
// Small kernels
// ---------------------------------------------------------------------------
__global__ void init_graph_kernel(float* indeg, float* loss)
{
    int i = blockIdx.x * blockDim.x + threadIdx.x;
    if (i < N_NODES) indeg[i] = 0.f;
    if (i == 0) *loss = 0.f;
}

__global__ void indeg_kernel(const int* __restrict__ dst, float* indeg)
{
    int e = blockIdx.x * blockDim.x + threadIdx.x;
    if (e < N_EDGES) atomicAdd(&indeg[dst[e]], 1.f);
}

__global__ __launch_bounds__(256)
void structure_kernel(const float* __restrict__ latp,
                      const float* __restrict__ W1, const float* __restrict__ b1,
                      const float* __restrict__ W2, const float* __restrict__ b2,
                      float* __restrict__ hlatp, float* __restrict__ hlatp2)
{
    __shared__ float sW1[256], sW2[256], sb1[16], sb2[16];
    int t = threadIdx.x;
    sW1[t] = W1[t]; sW2[t] = W2[t];
    if (t < 16) { sb1[t] = b1[t]; sb2[t] = b2[t]; }
    __syncthreads();
    int n = blockIdx.x * 256 + t;
    if (n >= N_NODES) return;
    float in[16], h[16];
#pragma unroll
    for (int c = 0; c < 16; ++c) in[c] = latp[(size_t)n * 16 + c];
#pragma unroll
    for (int j = 0; j < 16; ++j) {
        float s = sb1[j];
#pragma unroll
        for (int i = 0; i < 16; ++i) s += in[i] * sW1[i * 16 + j];
        h[j] = elu_f(s);
        hlatp[(size_t)n * 16 + j] = h[j];
    }
#pragma unroll
    for (int j = 0; j < 16; ++j) {
        float s = sb2[j];
#pragma unroll
        for (int i = 0; i < 16; ++i) s += h[i] * sW2[i * 16 + j];
        hlatp2[(size_t)n * 16 + j] = elu_f(s);
    }
}

__global__ void node_attn_kernel(const float* __restrict__ fs,
                                 const float* __restrict__ al, const float* __restrict__ ar,
                                 float* el, float* er, int F)
{
    int n = blockIdx.x * blockDim.x + threadIdx.x;
    if (n >= N_NODES) return;
    float sl = 0.f, sr = 0.f;
    for (int f = 0; f < F; ++f) {
        float lr = lrelu_f(fs[(size_t)n * F + f]);
        sl += lr * al[f];
        sr += lr * ar[f];
    }
    el[n] = sl; er[n] = sr;
}

__global__ void init_seg_kernel(float* mseg, float* zseg, float* rst, int rstlen)
{
    int i = blockIdx.x * blockDim.x + threadIdx.x;
    if (i < rstlen) rst[i] = 0.f;
    if (i < N_NODES) { mseg[i] = -INFINITY; zseg[i] = 0.f; }
}

template<int F>
__global__ void edge_logit_kernel(const int* __restrict__ src, const int* __restrict__ dst,
                                  const float* __restrict__ el, const float* __restrict__ er,
                                  const float* __restrict__ fs, const float* __restrict__ hl,
                                  const float* __restrict__ s_attn,
                                  const float* __restrict__ beta, const float* __restrict__ aw,
                                  float* logit, float* mseg)
{
    int e = blockIdx.x * blockDim.x + threadIdx.x;
    if (e >= N_EDGES) return;
    int s = src[e], d = dst[e];
    float att = el[s] + er[d];
    float se = 0.f, sds = 0.f;
#pragma unroll
    for (int c = 0; c < NCLASS; ++c) {
        float hs = hl[(size_t)s * NCLASS + c];
        float hd = hl[(size_t)d * NCLASS + c];
        se += hs * s_attn[c] * hd;
        float df = hs - hd;
        sds += df * df;
    }
    float sdf = 0.f;
#pragma unroll
    for (int f = 0; f < F; ++f) {
        float df = fs[(size_t)s * F + f] - fs[(size_t)d * F + f];
        sdf += df * df;
    }
    float a0 = aw[0], a1 = aw[1];
    float mx = fmaxf(a0, a1);
    float w0 = expf(a0 - mx), w1 = expf(a1 - mx);
    float inv = 1.f / (w0 + w1);
    w0 *= inv; w1 *= inv;
    float betaw = 2.f / (expf(-beta[0]) + 1.f);
    float lg = att + se - betaw * (w0 * sdf + w1 * sds);
    logit[e] = lg;
    atomicMaxF(&mseg[d], lg);
}

__global__ void edge_exp_kernel(const int* __restrict__ dst, float* logit,
                                const float* __restrict__ mseg, float* zseg)
{
    int e = blockIdx.x * blockDim.x + threadIdx.x;
    if (e >= N_EDGES) return;
    int d = dst[e];
    float a = expf(logit[e] - mseg[d]);
    logit[e] = a;
    atomicAdd(&zseg[d], a);
}

template<int F>
__global__ void edge_scatter_kernel(const int* __restrict__ src, const int* __restrict__ dst,
                                    const float* __restrict__ aval, const float* __restrict__ zseg,
                                    const float* __restrict__ fs, float* rst)
{
    long long i = (long long)blockIdx.x * blockDim.x + threadIdx.x;
    if (i >= (long long)N_EDGES * F) return;
    int e = (int)(i / F);
    int f = (int)(i % F);
    int s = src[e], d = dst[e];
    float coef = aval[e] / zseg[d];
    atomicAdd(&rst[(size_t)d * F + f], coef * fs[(size_t)s * F + f]);
}

template<int F, bool DO_ELU>
__global__ void combine_kernel(const float* __restrict__ rst, const float* __restrict__ fs,
                               const float* __restrict__ indeg, const float* __restrict__ eps,
                               float* out)
{
    long long i = (long long)blockIdx.x * blockDim.x + threadIdx.x;
    if (i >= (long long)N_NODES * F) return;
    int n = (int)(i / F);
    float perm = 1e-9f / (expf(-eps[0]) + 1.f);
    float v = rst[i] + fs[i] * (perm / (indeg[n] + 1e-9f));
    out[i] = DO_ELU ? elu_f(v) : v;
}

__global__ void output_kernel(const float* __restrict__ h2, float* out)
{
    int n = blockIdx.x * blockDim.x + threadIdx.x;
    if (n >= N_NODES) return;
    float v[16];
    float mx = -INFINITY;
#pragma unroll
    for (int c = 0; c < 16; ++c) {
        v[c] = elu_f(h2[(size_t)n * 16 + c]);
        mx = fmaxf(mx, v[c]);
    }
    float se = 0.f;
#pragma unroll
    for (int c = 0; c < 16; ++c) se += expf(v[c] - mx);
    float ls = mx + logf(se);
#pragma unroll
    for (int c = 0; c < 16; ++c) out[(size_t)n * 16 + c] = v[c] - ls;
}

__global__ __launch_bounds__(256)
void loss_kernel(const int* __restrict__ src, const int* __restrict__ dst,
                 const float* __restrict__ w, const float* __restrict__ hl2, float* loss)
{
    __shared__ float red[256];
    int t = threadIdx.x;
    int e = blockIdx.x * 256 + t;
    float c = 0.f;
    if (e < N_EDGES) {
        int s = src[e], d = dst[e];
        float dot = 0.f;
#pragma unroll
        for (int k = 0; k < 16; ++k)
            dot += hl2[(size_t)s * 16 + k] * hl2[(size_t)d * 16 + k];
        c = dot * w[e];
    }
    red[t] = c;
    __syncthreads();
    for (int off = 128; off > 0; off >>= 1) {
        if (t < off) red[t] += red[t + off];
        __syncthreads();
    }
    if (t == 0) atomicAdd(loss, red[0]);
}

// ---------------------------------------------------------------------------
extern "C" void kernel_launch(void* const* d_in, const int* in_sizes, int n_in,
                              void* d_out, int out_size, void* d_ws, size_t ws_size,
                              hipStream_t stream)
{
    const float* x       = (const float*)d_in[0];
    const float* weights = (const float*)d_in[1];
    const int*   src     = (const int*)  d_in[2];
    const int*   dst     = (const int*)  d_in[3];
    const float* latp    = (const float*)d_in[4];
    const float* sl_W    = (const float*)d_in[5];
    const float* sl_b    = (const float*)d_in[6];
    const float* slo_W   = (const float*)d_in[7];
    const float* slo_b   = (const float*)d_in[8];
    const float* fc1_W   = (const float*)d_in[9];
    const float* attn_l1 = (const float*)d_in[10];
    const float* attn_r1 = (const float*)d_in[11];
    const float* s_attn1 = (const float*)d_in[12];
    const float* beta1   = (const float*)d_in[13];
    const float* aw1     = (const float*)d_in[14];
    const float* eps1    = (const float*)d_in[15];
    const float* fc2_W   = (const float*)d_in[16];
    const float* attn_l2 = (const float*)d_in[17];
    const float* attn_r2 = (const float*)d_in[18];
    const float* s_attn2 = (const float*)d_in[19];
    const float* beta2   = (const float*)d_in[20];
    const float* aw2     = (const float*)d_in[21];
    const float* eps2    = (const float*)d_in[22];

    float* outp = (float*)d_out;
    float* loss = outp + (size_t)N_NODES * NCLASS;

    float* ws     = (float*)d_ws;
    float* fs1    = ws;                            ws += (size_t)N_NODES * NHID;
    float* rst1   = ws;                            ws += (size_t)N_NODES * NHID;
    float* h1     = ws;                            ws += (size_t)N_NODES * NHID;
    float* hlatp  = ws;                            ws += (size_t)N_NODES * NCLASS;
    float* hlatp2 = ws;                            ws += (size_t)N_NODES * NCLASS;
    float* fs2    = ws;                            ws += (size_t)N_NODES * NCLASS;
    float* rst2   = ws;                            ws += (size_t)N_NODES * NCLASS;
    float* indeg  = ws;                            ws += N_NODES;
    float* el     = ws;                            ws += N_NODES;
    float* er     = ws;                            ws += N_NODES;
    float* mseg   = ws;                            ws += N_NODES;
    float* zseg   = ws;                            ws += N_NODES;
    float* logit  = ws;                            ws += N_EDGES;
    bf16_t* wt1   = (bf16_t*)ws;                   // 32768 bf16
    bf16_t* wt2   = wt1 + (size_t)NFEAT * NHID;    // 1024 bf16

    const int TB = 256;
    const int gN   = (N_NODES + TB - 1) / TB;
    const int gE   = (N_EDGES + TB - 1) / TB;
    const int gM   = (N_NODES + 127) / 128;
    const int gN64 = (N_NODES * NHID + TB - 1) / TB;
    const int gN16 = (N_NODES * NCLASS + TB - 1) / TB;
    const int gE64 = (int)(((long long)N_EDGES * NHID + TB - 1) / TB);
    const int gE16 = (int)(((long long)N_EDGES * NCLASS + TB - 1) / TB);
    const int gW   = (NFEAT * NHID + TB - 1) / TB;

    // graph prep
    init_graph_kernel<<<gN, TB, 0, stream>>>(indeg, loss);
    indeg_kernel<<<gE, TB, 0, stream>>>(dst, indeg);
    prep_weights_kernel<<<gW, TB, 0, stream>>>(fc1_W, fc2_W, wt1, wt2);
    structure_kernel<<<gN, TB, 0, stream>>>(latp, sl_W, sl_b, slo_W, slo_b, hlatp, hlatp2);

    // ------------------- layer 1 -------------------
    gemm_wmma_kernel<NFEAT, 4><<<gM, TB, 0, stream>>>(x, wt1, fs1);
    node_attn_kernel<<<gN, TB, 0, stream>>>(fs1, attn_l1, attn_r1, el, er, NHID);
    init_seg_kernel<<<gN64, TB, 0, stream>>>(mseg, zseg, rst1, N_NODES * NHID);
    edge_logit_kernel<NHID><<<gE, TB, 0, stream>>>(src, dst, el, er, fs1, hlatp,
                                                   s_attn1, beta1, aw1, logit, mseg);
    edge_exp_kernel<<<gE, TB, 0, stream>>>(dst, logit, mseg, zseg);
    edge_scatter_kernel<NHID><<<gE64, TB, 0, stream>>>(src, dst, logit, zseg, fs1, rst1);
    combine_kernel<NHID, true><<<gN64, TB, 0, stream>>>(rst1, fs1, indeg, eps1, h1);

    // ------------------- layer 2 -------------------
    gemm_wmma_kernel<NHID, 1><<<gM, TB, 0, stream>>>(h1, wt2, fs2);
    node_attn_kernel<<<gN, TB, 0, stream>>>(fs2, attn_l2, attn_r2, el, er, NCLASS);
    init_seg_kernel<<<gN16, TB, 0, stream>>>(mseg, zseg, rst2, N_NODES * NCLASS);
    edge_logit_kernel<NCLASS><<<gE, TB, 0, stream>>>(src, dst, el, er, fs2, hlatp2,
                                                     s_attn2, beta2, aw2, logit, mseg);
    edge_exp_kernel<<<gE, TB, 0, stream>>>(dst, logit, mseg, zseg);
    edge_scatter_kernel<NCLASS><<<gE16, TB, 0, stream>>>(src, dst, logit, zseg, fs2, rst2);
    combine_kernel<NCLASS, false><<<gN16, TB, 0, stream>>>(rst2, fs2, indeg, eps2, rst2);

    // ------------------- outputs -------------------
    output_kernel<<<gN, TB, 0, stream>>>(rst2, outp);
    loss_kernel<<<gE, TB, 0, stream>>>(src, dst, weights, hlatp2, loss);
}
